// Encoder_18313740550502
// MI455X (gfx1250) — matrix operand order repaired and checked by reading, not compiled, not run on hardware
//
#include <hip/hip_runtime.h>
#include <hip/hip_bf16.h>
#include <stdint.h>

// ---------------- problem dims ----------------
constexpr int cB = 16, cS = 512, cD = 1024, cH = 16, cF = 4096;
constexpr int cND = 4, cL = 4, cHD = 64;
constexpr int cM = cB * cS;          // 8192 tokens

// ---------------- WMMA types ----------------
typedef __bf16 bf16_t;
typedef bf16_t v16bf __attribute__((ext_vector_type(16)));
typedef float  v8f   __attribute__((ext_vector_type(8)));

__device__ __forceinline__ uint16_t f2bf(float x) {
  uint32_t u = __float_as_uint(x);
  return (uint16_t)((u + 0x7FFFu + ((u >> 16) & 1u)) >> 16);  // RNE truncate
}

struct BF16Frag { union { uint32_t u[8]; v16bf v; }; };

__device__ __forceinline__ v8f v8f_zero() {
  v8f z;
#pragma unroll
  for (int i = 0; i < 8; ++i) z[i] = 0.0f;
  return z;
}

// A fragment: 16x32 bf16 tile of row-major A[row, k] (lda elements).
// ISA layout: lanes 0-15 / 16-31 both hold M=lane&15; K pairs:
// VGPR0-3 -> K in [0,16) (lo-half lanes K=0..7, hi-half K=8..15),
// VGPR4-7 -> K in [16,32) likewise.
__device__ __forceinline__ void load_a_frag(BF16Frag& f, const uint16_t* A, int lda,
                                            int row_base, int k0, int lane) {
  const int m   = lane & 15;
  const int hi8 = (lane >> 4) << 3;  // 0 or 8
  const uint16_t* rp = A + (size_t)(row_base + m) * lda + k0 + hi8;
#pragma unroll
  for (int i = 0; i < 4; ++i) f.u[i]     = *(const uint32_t*)(rp + 2 * i);
#pragma unroll
  for (int i = 0; i < 4; ++i) f.u[4 + i] = *(const uint32_t*)(rp + 16 + 2 * i);
}

// B fragment: 32x16 bf16 tile. Element (K,N) lives at Bt[(col_base+N)*ldn + K]
// (K always contiguous: all operands are pre-transposed so B loads are dwords).
// Lanes 0-15: N=lane, K=0..7,16..23; lanes 16-31: N=lane-16, K=8..15,24..31 (same K order as A).
__device__ __forceinline__ void load_b_frag(BF16Frag& f, const uint16_t* Bt, int ldn,
                                            int col_base, int k0, int lane) {
  const int n    = lane & 15;
  const int hi8  = (lane >> 4) << 3;  // 0 or 8
  const uint16_t* cp = Bt + (size_t)(col_base + n) * ldn + (k0 + hi8);
#pragma unroll
  for (int v = 0; v < 8; ++v) f.u[v] = *(const uint32_t*)(cp + 2 * v + (v >= 4 ? 8 : 0));
}

// ---------------- generic batched wave-tiled WMMA GEMM ----------------
// Wave tile: 32 rows x 64 cols (2 A frags, 4 B frags, 8 WMMA per K-step).
// Block: 4 waves -> 128 rows x 64 cols.
// out = Cin + rowscale[row]*alpha*( relu?( A@Bt + bias ) )
struct GemmP {
  const uint16_t* A;  int lda;  long long strideAz;
  const uint16_t* Bt; int ldn;  long long strideBz;
  const float* bias;                     // per-column [N] or null
  const float* rowscale; int rsStride;   // null -> 1.0
  float alpha;
  const float* Cin; int ldci; long long strideCiz;  // null -> 0
  float* outF; uint16_t* outBF; int ldc; long long strideCz;
  int M, N, K, relu;
};

__global__ void __launch_bounds__(128) wgemm_kernel(GemmP p) {
  const int lane   = threadIdx.x & 31;
  const int wave   = threadIdx.x >> 5;
  const int z      = blockIdx.z;
  const int tile_m = blockIdx.x * 128 + wave * 32;
  const int tile_n = blockIdx.y * 64;
  if (tile_m >= p.M || tile_n >= p.N) return;
  const uint16_t* A  = p.A  + (size_t)z * p.strideAz;
  const uint16_t* Bt = p.Bt + (size_t)z * p.strideBz;
  v8f acc[2][4];
#pragma unroll
  for (int i = 0; i < 2; ++i)
#pragma unroll
    for (int j = 0; j < 4; ++j) acc[i][j] = v8f_zero();

  for (int k0 = 0; k0 < p.K; k0 += 32) {
    // prefetch next K-tile (speculative, dropped if OOB)
    __builtin_prefetch(A  + (size_t)(tile_m + (lane & 15)) * p.lda + k0 + 32, 0, 1);
    __builtin_prefetch(Bt + (size_t)(tile_n + (lane & 15)) * p.ldn + k0 + 32, 0, 1);
    BF16Frag a0, a1;
    load_a_frag(a0, A, p.lda, tile_m,      k0, lane);
    load_a_frag(a1, A, p.lda, tile_m + 16, k0, lane);
#pragma unroll
    for (int j = 0; j < 4; ++j) {
      BF16Frag b; load_b_frag(b, Bt, p.ldn, tile_n + j * 16, k0, lane);
      acc[0][j] = __builtin_amdgcn_wmma_f32_16x16x32_bf16(false, a0.v, false, b.v,
                                                          (short)0, acc[0][j], false, false);
      acc[1][j] = __builtin_amdgcn_wmma_f32_16x16x32_bf16(false, a1.v, false, b.v,
                                                          (short)0, acc[1][j], false, false);
    }
  }
  const int m8  = ((lane >> 4) & 1) * 8;
  const int nlo = lane & 15;
#pragma unroll
  for (int i = 0; i < 2; ++i) {
#pragma unroll
    for (int r = 0; r < 8; ++r) {
      const int row = tile_m + i * 16 + r + m8;
      const float rs = (p.rowscale ? p.rowscale[(size_t)row * p.rsStride] : 1.0f) * p.alpha;
#pragma unroll
      for (int j = 0; j < 4; ++j) {
        const int col = tile_n + j * 16 + nlo;
        float t = acc[i][j][r] + (p.bias ? p.bias[col] : 0.0f);
        if (p.relu) t = fmaxf(t, 0.0f);
        t *= rs;
        if (p.Cin) t += p.Cin[(size_t)z * p.strideCiz + (size_t)row * p.ldci + col];
        const size_t co = (size_t)z * p.strideCz + (size_t)row * p.ldc + col;
        if (p.outF)  p.outF[co]  = t;
        if (p.outBF) p.outBF[co] = f2bf(t);
      }
    }
  }
}

// ---------------- fused domain-mix GEMM (QKV) ----------------
// out[t,e] = outscale * sum_n route[t,n] * ( (X@W_n)[t,e] + bias[n,e] )
struct MixP {
  const uint16_t* X;    // [M,K] bf16
  const uint16_t* Wt;   // [ND][N][K] bf16 (transposed weights)
  const float* bias;    // [ND][N]
  const float* route;   // [M][ND]
  float outscale;
  uint16_t* outBF;      // [M,N]
  int M, N, K;
};

__global__ void __launch_bounds__(128) mix_gemm_kernel(MixP p) {
  const int lane   = threadIdx.x & 31;
  const int wave   = threadIdx.x >> 5;
  const int tile_m = blockIdx.x * 64 + wave * 16;
  const int tile_n = blockIdx.y * 64;
  const int m8  = ((lane >> 4) & 1) * 8;
  const int nlo = lane & 15;
  v8f fin[4];
#pragma unroll
  for (int j = 0; j < 4; ++j) fin[j] = v8f_zero();
  for (int n = 0; n < cND; ++n) {
    const uint16_t* Bt = p.Wt + (size_t)n * p.N * p.K;
    v8f acc[4];
#pragma unroll
    for (int j = 0; j < 4; ++j) acc[j] = v8f_zero();
    for (int k0 = 0; k0 < p.K; k0 += 32) {
      __builtin_prefetch(Bt + (size_t)(tile_n + (lane & 15)) * p.K + k0 + 32, 0, 1);
      BF16Frag a; load_a_frag(a, p.X, p.K, tile_m, k0, lane);
#pragma unroll
      for (int j = 0; j < 4; ++j) {
        BF16Frag b; load_b_frag(b, Bt, p.K, tile_n + j * 16, k0, lane);
        acc[j] = __builtin_amdgcn_wmma_f32_16x16x32_bf16(false, a.v, false, b.v,
                                                         (short)0, acc[j], false, false);
      }
    }
#pragma unroll
    for (int r = 0; r < 8; ++r) {
      const int row = tile_m + r + m8;
      const float dv = p.route[(size_t)row * cND + n];
#pragma unroll
      for (int j = 0; j < 4; ++j) {
        const int col = tile_n + j * 16 + nlo;
        fin[j][r] += dv * (acc[j][r] + p.bias[(size_t)n * p.N + col]);
      }
    }
  }
#pragma unroll
  for (int r = 0; r < 8; ++r) {
    const int row = tile_m + r + m8;
#pragma unroll
    for (int j = 0; j < 4; ++j) {
      const int col = tile_n + j * 16 + nlo;
      p.outBF[(size_t)row * p.N + col] = f2bf(fin[j][r] * p.outscale);
    }
  }
}

// ---------------- small support kernels ----------------
__global__ void embed_kernel(const int* src, const float* tok, const float* pos,
                             float* x, uint16_t* xbf) {
  size_t idx = (size_t)blockIdx.x * blockDim.x + threadIdx.x;
  if (idx >= (size_t)cM * cD) return;
  int d = (int)(idx % cD);
  size_t t = idx / cD;
  int s = (int)(t % cS);
  float v = tok[(size_t)src[t] * cD + d] * 32.0f /*sqrt(1024)*/ + pos[(size_t)s * cD + d];
  x[idx] = v;
  xbf[idx] = f2bf(v);
}

// W [slice][K][N] fp32 -> Wt [slice][N][K] bf16 (slice = blockIdx.y)
__global__ void transpose_bf16_kernel(const float* W, uint16_t* Wt, int K, int N) {
  const size_t total = (size_t)K * N;
  size_t idx = (size_t)blockIdx.x * blockDim.x + threadIdx.x;
  if (idx >= total) return;
  const size_t off = (size_t)blockIdx.y * total;
  int k = (int)(idx % K);
  int n = (int)(idx / K);
  Wt[off + idx] = f2bf(W[off + (size_t)k * N + n]);
}

// V [B][S][D] bf16 -> Vt [B][D][S] bf16  (so PV GEMM B-fragments load contiguously)
__global__ void transpose_v_kernel(const uint16_t* V, uint16_t* Vt) {
  size_t idx = (size_t)blockIdx.x * blockDim.x + threadIdx.x;
  if (idx >= (size_t)cB * cS * cD) return;
  int e = (int)(idx % cD);
  size_t t = idx / cD;
  int j = (int)(t % cS);
  int b = (int)(t / cS);
  Vt[((size_t)b * cD + e) * cS + j] = V[idx];
}

// per-token router: d = (1-eps)*softmax(x@Wr + br) + eps/ND ; optional domsum += 0.5*d
__global__ void router_kernel(const float* x, const float* Wr, const float* br,
                              float* dout, float* domsum) {
  int t = blockIdx.x * blockDim.x + threadIdx.x;
  if (t >= cM) return;
  const float* xr = x + (size_t)t * cD;
  float a0 = br[0], a1 = br[1], a2 = br[2], a3 = br[3];
  for (int d = 0; d < cD; ++d) {
    float xv = xr[d];
    a0 += xv * Wr[d * 4 + 0];
    a1 += xv * Wr[d * 4 + 1];
    a2 += xv * Wr[d * 4 + 2];
    a3 += xv * Wr[d * 4 + 3];
  }
  float mx = fmaxf(fmaxf(a0, a1), fmaxf(a2, a3));
  float e0 = __expf(a0 - mx), e1 = __expf(a1 - mx), e2 = __expf(a2 - mx), e3 = __expf(a3 - mx);
  float inv = 1.0f / (e0 + e1 + e2 + e3);
  float d0 = 0.9f * e0 * inv + 0.025f;
  float d1 = 0.9f * e1 * inv + 0.025f;
  float d2 = 0.9f * e2 * inv + 0.025f;
  float d3 = 0.9f * e3 * inv + 0.025f;
  size_t o = (size_t)t * 4;
  dout[o + 0] = d0; dout[o + 1] = d1; dout[o + 2] = d2; dout[o + 3] = d3;
  if (domsum) {
    domsum[o + 0] += 0.5f * d0; domsum[o + 1] += 0.5f * d1;
    domsum[o + 2] += 0.5f * d2; domsum[o + 3] += 0.5f * d3;
  }
}

// softmax over S=512 scores (fp32 in, bf16 out); mask[j]==0 -> -1e10
__global__ void __launch_bounds__(256) softmax_kernel(const float* scores, uint16_t* probs,
                                                      const int* mask) {
  const int h = blockIdx.y, i = blockIdx.x, t = threadIdx.x;
  const float* row = scores + ((size_t)h * cS + i) * cS;
  uint16_t* prow   = probs  + ((size_t)h * cS + i) * cS;
  float v0 = (mask[t]       != 0) ? row[t]       : -1e10f;
  float v1 = (mask[t + 256] != 0) ? row[t + 256] : -1e10f;
  __shared__ float red[256];
  red[t] = fmaxf(v0, v1); __syncthreads();
  for (int s2 = 128; s2 > 0; s2 >>= 1) {
    if (t < s2) red[t] = fmaxf(red[t], red[t + s2]);
    __syncthreads();
  }
  const float rmax = red[0]; __syncthreads();
  float e0 = __expf(v0 - rmax), e1 = __expf(v1 - rmax);
  red[t] = e0 + e1; __syncthreads();
  for (int s2 = 128; s2 > 0; s2 >>= 1) {
    if (t < s2) red[t] += red[t + s2];
    __syncthreads();
  }
  const float inv = 1.0f / red[0];
  prow[t]       = f2bf(e0 * inv);
  prow[t + 256] = f2bf(e1 * inv);
}

// LayerNorm over D=1024 of s; writes x (f32) and xbf (bf16)
__global__ void __launch_bounds__(256) ln_kernel(const float* s, const float* g, const float* bta,
                                                 float* x, uint16_t* xbf) {
  const int t = blockIdx.x;
  const float* row = s + (size_t)t * cD;
  __shared__ float r1[256], r2[256];
  float ls = 0.f, lq = 0.f;
  for (int i = threadIdx.x; i < cD; i += 256) { float v = row[i]; ls += v; lq += v * v; }
  r1[threadIdx.x] = ls; r2[threadIdx.x] = lq; __syncthreads();
  for (int st = 128; st > 0; st >>= 1) {
    if (threadIdx.x < st) { r1[threadIdx.x] += r1[threadIdx.x + st]; r2[threadIdx.x] += r2[threadIdx.x + st]; }
    __syncthreads();
  }
  const float mu   = r1[0] * (1.0f / cD);
  const float var  = r2[0] * (1.0f / cD) - mu * mu;
  const float rstd = rsqrtf(var + 1e-5f);
  for (int i = threadIdx.x; i < cD; i += 256) {
    float y = (row[i] - mu) * rstd * g[i] + bta[i];
    x[(size_t)t * cD + i]   = y;
    xbf[(size_t)t * cD + i] = f2bf(y);
  }
}

__global__ void out_kernel(const float* x, const float* domsum, float* out) {
  const size_t nx  = (size_t)cM * cD;
  const size_t ndm = (size_t)cM * cND;
  size_t idx = (size_t)blockIdx.x * blockDim.x + threadIdx.x;
  if (idx < nx) out[idx] = x[idx];
  else if (idx < nx + ndm) out[idx] = domsum[idx - nx] * (1.0f / cL);
}

// ---------------- host orchestration ----------------
extern "C" void kernel_launch(void* const* d_in, const int* in_sizes, int n_in,
                              void* d_out, int out_size, void* d_ws, size_t ws_size,
                              hipStream_t stream) {
  (void)in_sizes; (void)n_in; (void)out_size; (void)ws_size;
  // inputs in setup_inputs() order
  const int*   src      = (const int*)d_in[0];
  const int*   src_mask = (const int*)d_in[1];
  const float* tok_emb  = (const float*)d_in[2];
  const float* pos_emb  = (const float*)d_in[3];
  const float* Wrq = (const float*)d_in[4],  *brq = (const float*)d_in[5];
  const float* Wrk = (const float*)d_in[6],  *brk = (const float*)d_in[7];
  const float* Wrv = (const float*)d_in[8],  *brv = (const float*)d_in[9];
  const float* Wrf = (const float*)d_in[10], *brf = (const float*)d_in[11];
  const float* Wq  = (const float*)d_in[12], *bq  = (const float*)d_in[13];
  const float* Wk  = (const float*)d_in[14], *bk  = (const float*)d_in[15];
  const float* Wv  = (const float*)d_in[16], *bv  = (const float*)d_in[17];
  const float* Wo  = (const float*)d_in[18], *bo  = (const float*)d_in[19];
  const float* W1  = (const float*)d_in[20], *b1  = (const float*)d_in[21];
  const float* W2  = (const float*)d_in[22], *b2  = (const float*)d_in[23];
  const float* g1  = (const float*)d_in[24], *be1 = (const float*)d_in[25];
  const float* g2  = (const float*)d_in[26], *be2 = (const float*)d_in[27];

  // scratch carve
  char* w = (char*)d_ws;
  auto carve = [&](size_t bytes) { char* p = w; w += (bytes + 255) & ~(size_t)255; return p; };
  float*    x      = (float*)   carve((size_t)cM * cD * 4);
  float*    tmp    = (float*)   carve((size_t)cM * cD * 4);
  uint16_t* xbf    = (uint16_t*)carve((size_t)cM * cD * 2);
  uint16_t* qbf    = (uint16_t*)carve((size_t)cM * cD * 2);
  uint16_t* kbf    = (uint16_t*)carve((size_t)cM * cD * 2);
  uint16_t* vbf    = (uint16_t*)carve((size_t)cM * cD * 2);
  uint16_t* vt     = (uint16_t*)carve((size_t)cM * cD * 2);
  uint16_t* obf    = (uint16_t*)carve((size_t)cM * cD * 2);
  uint16_t* hbf    = (uint16_t*)carve((size_t)cM * cF * 2);
  uint16_t* wt     = (uint16_t*)carve((size_t)cND * cD * cD * 2);  // also fits D*F
  float*    scores = (float*)   carve((size_t)cH * cS * cS * 4);
  uint16_t* probs  = (uint16_t*)carve((size_t)cH * cS * cS * 2);
  float*    dq     = (float*)   carve((size_t)cM * cND * 4);
  float*    dk     = (float*)   carve((size_t)cM * cND * 4);
  float*    dv     = (float*)   carve((size_t)cM * cND * 4);
  float*    dff    = (float*)   carve((size_t)cM * cND * 4);
  float*    domsum = (float*)   carve((size_t)cM * cND * 4);

  hipMemsetAsync(domsum, 0, (size_t)cM * cND * 4, stream);

  const int TPB = 256;
  embed_kernel<<<((size_t)cM * cD + TPB - 1) / TPB, TPB, 0, stream>>>(src, tok_emb, pos_emb, x, xbf);

  for (int l = 0; l < cL; ++l) {
    // routers
    router_kernel<<<cM / TPB, TPB, 0, stream>>>(x, Wrq + (size_t)l * cD * cND, brq + l * cND, dq, domsum);
    router_kernel<<<cM / TPB, TPB, 0, stream>>>(x, Wrk + (size_t)l * cD * cND, brk + l * cND, dk, nullptr);
    router_kernel<<<cM / TPB, TPB, 0, stream>>>(x, Wrv + (size_t)l * cD * cND, brv + l * cND, dv, nullptr);

    // Q/K/V domain-mix GEMMs (fused over ND), out = bf16
    const float* Wqkv[3] = {Wq, Wk, Wv};
    const float* bqkv[3] = {bq, bk, bv};
    const float* dqkv[3] = {dq, dk, dv};
    uint16_t*    oqkv[3] = {qbf, kbf, vbf};
    for (int m = 0; m < 3; ++m) {
      transpose_bf16_kernel<<<dim3(((size_t)cD * cD + TPB - 1) / TPB, cND), TPB, 0, stream>>>(
          Wqkv[m] + (size_t)l * cND * cD * cD, wt, cD, cD);
      MixP mp{};
      mp.X = xbf; mp.Wt = wt;
      mp.bias = bqkv[m] + (size_t)l * cND * cD;
      mp.route = dqkv[m];
      mp.outscale = 1.0f / cND;
      mp.outBF = oqkv[m];
      mp.M = cM; mp.N = cD; mp.K = cD;
      mix_gemm_kernel<<<dim3(cM / 64, cD / 64), 128, 0, stream>>>(mp);
    }

    // V transpose for the PV GEMM (contiguous B-fragment loads)
    transpose_v_kernel<<<((size_t)cM * cD + TPB - 1) / TPB, TPB, 0, stream>>>(vbf, vt);

    // attention, per batch element (keeps score scratch small)
    for (int b = 0; b < cB; ++b) {
      GemmP sp{};
      sp.A = qbf + (size_t)b * cS * cD;  sp.lda = cD; sp.strideAz = cHD;
      sp.Bt = kbf + (size_t)b * cS * cD; sp.ldn = cD; sp.strideBz = cHD;
      sp.bias = nullptr; sp.rowscale = nullptr; sp.rsStride = 0;
      sp.alpha = 0.125f;  // 1/sqrt(HD)
      sp.Cin = nullptr; sp.ldci = 0; sp.strideCiz = 0;
      sp.outF = scores; sp.outBF = nullptr; sp.ldc = cS; sp.strideCz = (long long)cS * cS;
      sp.M = cS; sp.N = cS; sp.K = cHD; sp.relu = 0;
      wgemm_kernel<<<dim3(cS / 128, cS / 64, cH), 128, 0, stream>>>(sp);

      softmax_kernel<<<dim3(cS, cH), 256, 0, stream>>>(scores, probs, src_mask + (size_t)b * cS);

      GemmP ap{};
      ap.A = probs; ap.lda = cS; ap.strideAz = (long long)cS * cS;
      ap.Bt = vt + (size_t)b * cD * cS; ap.ldn = cS; ap.strideBz = (long long)cHD * cS;
      ap.bias = nullptr; ap.rowscale = nullptr; ap.rsStride = 0; ap.alpha = 1.0f;
      ap.Cin = nullptr; ap.ldci = 0; ap.strideCiz = 0;
      ap.outF = nullptr; ap.outBF = obf + (size_t)b * cS * cD; ap.ldc = cD; ap.strideCz = cHD;
      ap.M = cS; ap.N = cHD; ap.K = cS; ap.relu = 0;
      wgemm_kernel<<<dim3(cS / 128, 1, cH), 128, 0, stream>>>(ap);
    }

    // output projection + residual into tmp, then LN1
    transpose_bf16_kernel<<<dim3(((size_t)cD * cD + TPB - 1) / TPB, 1), TPB, 0, stream>>>(
        Wo + (size_t)l * cD * cD, wt, cD, cD);
    {
      GemmP op{};
      op.A = obf; op.lda = cD; op.strideAz = 0;
      op.Bt = wt; op.ldn = cD; op.strideBz = 0;
      op.bias = bo + (size_t)l * cD; op.rowscale = nullptr; op.rsStride = 0; op.alpha = 1.0f;
      op.Cin = x; op.ldci = cD; op.strideCiz = 0;
      op.outF = tmp; op.outBF = nullptr; op.ldc = cD; op.strideCz = 0;
      op.M = cM; op.N = cD; op.K = cD; op.relu = 0;
      wgemm_kernel<<<dim3(cM / 128, cD / 64, 1), 128, 0, stream>>>(op);
    }
    ln_kernel<<<cM, 256, 0, stream>>>(tmp, g1 + (size_t)l * cD, be1 + (size_t)l * cD, x, xbf);

    // FFN router (contributes to domsum)
    router_kernel<<<cM / TPB, TPB, 0, stream>>>(x, Wrf + (size_t)l * cD * cND, brf + l * cND, dff, domsum);

    // FFN: per-domain W1(+relu) -> hbf, then W2 row-scaled accumulate into tmp (residual from x)
    for (int n = 0; n < cND; ++n) {
      transpose_bf16_kernel<<<dim3(((size_t)cD * cF + TPB - 1) / TPB, 1), TPB, 0, stream>>>(
          W1 + ((size_t)l * cND + n) * cD * cF, wt, cD, cF);
      GemmP f1{};
      f1.A = xbf; f1.lda = cD; f1.strideAz = 0;
      f1.Bt = wt; f1.ldn = cD; f1.strideBz = 0;
      f1.bias = b1 + ((size_t)l * cND + n) * cF;
      f1.rowscale = nullptr; f1.rsStride = 0; f1.alpha = 1.0f;
      f1.Cin = nullptr; f1.ldci = 0; f1.strideCiz = 0;
      f1.outF = nullptr; f1.outBF = hbf; f1.ldc = cF; f1.strideCz = 0;
      f1.M = cM; f1.N = cF; f1.K = cD; f1.relu = 1;
      wgemm_kernel<<<dim3(cM / 128, cF / 64, 1), 128, 0, stream>>>(f1);

      transpose_bf16_kernel<<<dim3(((size_t)cF * cD + TPB - 1) / TPB, 1), TPB, 0, stream>>>(
          W2 + ((size_t)l * cND + n) * cF * cD, wt, cF, cD);
      GemmP f2{};
      f2.A = hbf; f2.lda = cF; f2.strideAz = 0;
      f2.Bt = wt; f2.ldn = cF; f2.strideBz = 0;
      f2.bias = b2 + ((size_t)l * cND + n) * cD;
      f2.rowscale = dff + n; f2.rsStride = cND; f2.alpha = 1.0f;
      f2.Cin = (n == 0) ? x : tmp; f2.ldci = cD; f2.strideCiz = 0;
      f2.outF = tmp; f2.outBF = nullptr; f2.ldc = cD; f2.strideCz = 0;
      f2.M = cM; f2.N = cD; f2.K = cF; f2.relu = 0;
      wgemm_kernel<<<dim3(cM / 128, cD / 64, 1), 128, 0, stream>>>(f2);
    }
    ln_kernel<<<cM, 256, 0, stream>>>(tmp, g2 + (size_t)l * cD, be2 + (size_t)l * cD, x, xbf);
  }

  const size_t nout = (size_t)cM * cD + (size_t)cM * cND;
  out_kernel<<<(nout + TPB - 1) / TPB, TPB, 0, stream>>>(x, domsum, (float*)d_out);
}
